// TennisModelGATTransformer_38079180046438
// MI455X (gfx1250) — compile-verified, matrix operand order checked by reading, not run on hardware
//
#include <hip/hip_runtime.h>
#include <hip/hip_bf16.h>

typedef __attribute__((ext_vector_type(16))) _Float16 v16h;
typedef __attribute__((ext_vector_type(8)))  _Float16 v8h;
typedef __attribute__((ext_vector_type(8)))  float    v8f;
typedef __attribute__((ext_vector_type(4)))  float    v4f;

// ---------------- model constants ----------------
#define NN   50000
#define EE   800000
#define FIN  32
#define HID  64
#define TD   64
#define NH   8
#define HD   8
#define NL   4
#define FFD  2048
#define BB   128
#define LL   128
#define SS   256      // 2*B sequences
#define RR   32768    // SS*LL rows
#define MF   36
#define CD   64

// ---------------- WMMA helpers ----------------
__device__ inline v8f wmma_f16(v16h a, v16h b, v8f c) {
    // emits v_wmma_f32_16x16x32_f16
    return __builtin_amdgcn_wmma_f32_16x16x32_f16(false, a, false, b, (short)0, c, false, false);
}

// Build a 16x32 f16 fragment from row-major f32 source (K must be mult of 32).
// ISA layout (16-bit A 16x32): lanes 0-15 -> row=lane, K {0..7, 16..23};
// lanes 16-31 -> row=lane-16, K {8..15, 24..31}. B (KxN) uses the symmetric
// layout with lane indexing the N column of W (row of W since B = W^T).
// All chunks are 16-byte aligned -> 4x vector loads, no predication.
__device__ inline v16h frag_f32(const float* __restrict__ p, int ld,
                                int row0, int k0, int lane) {
    int r  = lane & 15;
    int kb = (lane & 16) ? 8 : 0;
    const float* q = p + (size_t)(row0 + r) * ld + k0 + kb;
    v4f q0 = *(const v4f*)(q);
    v4f q1 = *(const v4f*)(q + 4);
    v4f q2 = *(const v4f*)(q + 16);
    v4f q3 = *(const v4f*)(q + 20);
    v16h f;
#pragma unroll
    for (int i = 0; i < 4; ++i) {
        f[i]      = (_Float16)q0[i];
        f[4 + i]  = (_Float16)q1[i];
        f[8 + i]  = (_Float16)q2[i];
        f[12 + i] = (_Float16)q3[i];
    }
    return f;
}

// Same but from f16 storage (LDS staging); two aligned 16B loads.
__device__ inline v16h frag_f16(const _Float16* __restrict__ p, int ld,
                                int row0, int k0, int lane) {
    int r  = lane & 15;
    int kb = (lane & 16) ? 8 : 0;
    const _Float16* q = p + (size_t)(row0 + r) * ld + k0 + kb;
    v8h lo = *(const v8h*)(q);
    v8h hi = *(const v8h*)(q + 16);
    v16h f;
#pragma unroll
    for (int i = 0; i < 8; ++i) { f[i] = lo[i]; f[8 + i] = hi[i]; }
    return f;
}

// ---------------- generic WMMA GEMM: C[M,N] = A[M,K] @ W[N,K]^T (+bias) ----
// one wave per 16x16 tile; 256 threads = 8 waves per block; K mult of 32
__global__ __launch_bounds__(256) void gemm_wmma(
    const float* __restrict__ A, const float* __restrict__ W,
    const float* __restrict__ bias, float* __restrict__ C,
    int M, int Nout, int K, int tilesN, int totalTiles) {
    int wave = threadIdx.x >> 5;
    int lane = threadIdx.x & 31;
    int tile = blockIdx.x * 8 + wave;
    if (tile >= totalTiles) return;           // wave-uniform guard: EXEC all-ones inside
    int tm = tile / tilesN;
    int tn = tile - tm * tilesN;
    v8f c = {};
    for (int k0 = 0; k0 < K; k0 += 32) {
        v16h a = frag_f32(A, K, tm * 16, k0, lane);
        v16h b = frag_f32(W, K, tn * 16, k0, lane);
        c = wmma_f16(a, b, c);
    }
    int col   = tn * 16 + (lane & 15);
    int row0  = tm * 16 + ((lane & 16) ? 8 : 0);
    float bv  = bias ? bias[col] : 0.f;
#pragma unroll
    for (int i = 0; i < 8; ++i)
        C[(size_t)(row0 + i) * Nout + col] = c[i] + bv;
}

// ---------------- fused FFN: out = relu(h@W1^T+b1)@W2^T+b2 ------------------
// block = 16 rows; ReLU intermediate (16x2048) staged in LDS as f16 (66 KB)
#define MIDLD 2064  // 2048 + 16 pad (row stride 4128 B, 16B-aligned)
__global__ __launch_bounds__(256) void ffn_fused(
    const float* __restrict__ h,  const float* __restrict__ w1,
    const float* __restrict__ b1, const float* __restrict__ w2,
    const float* __restrict__ b2, float* __restrict__ out) {
    extern __shared__ _Float16 mid[];   // [16][MIDLD]
    int wave = threadIdx.x >> 5;
    int lane = threadIdx.x & 31;
    int m0 = blockIdx.x * 16;
    // A fragments (16x64) reused for every N tile of FF1
    v16h a0 = frag_f32(h, TD, m0, 0, lane);
    v16h a1 = frag_f32(h, TD, m0, 32, lane);
    for (int nt = wave; nt < FFD / 16; nt += 8) {
        v8f c = {};
        v16h bf0 = frag_f32(w1, TD, nt * 16, 0, lane);
        v16h bf1 = frag_f32(w1, TD, nt * 16, 32, lane);
        c = wmma_f16(a0, bf0, c);
        c = wmma_f16(a1, bf1, c);
        int col = nt * 16 + (lane & 15);
        int r0  = (lane & 16) ? 8 : 0;
        float bv = b1[col];
#pragma unroll
        for (int i = 0; i < 8; ++i) {
            float v = c[i] + bv;
            mid[(size_t)(r0 + i) * MIDLD + col] = (_Float16)(v > 0.f ? v : 0.f);
        }
    }
    __syncthreads();
    if (wave < 4) {                          // 4 output tiles of 16 cols (TD=64)
        v8f c = {};
        for (int k0 = 0; k0 < FFD; k0 += 32) {
            v16h a = frag_f16(mid, MIDLD, 0, k0, lane);
            v16h b = frag_f32(w2, FFD, wave * 16, k0, lane);
            c = wmma_f16(a, b, c);
        }
        int col = wave * 16 + (lane & 15);
        int r0  = (lane & 16) ? 8 : 0;
        float bv = b2[col];
#pragma unroll
        for (int i = 0; i < 8; ++i)
            out[(size_t)(m0 + r0 + i) * TD + col] = c[i] + bv;
    }
}

// ---------------- attention (HD=8 -> VALU, LDS score staging) ---------------
__global__ __launch_bounds__(128) void attn_kernel(
    const float* __restrict__ qkv, const int* __restrict__ p1_len,
    const int* __restrict__ p2_len, float* __restrict__ o) {
    extern __shared__ float sc[];       // [128][128]
    int bh = blockIdx.x;
    int b  = bh >> 3;
    int hh = bh & 7;
    int len = (b < BB) ? p1_len[b] : p2_len[b - BB];
    int q = threadIdx.x;
    const float* qp = qkv + ((size_t)(b * LL + q)) * (3 * TD) + hh * HD;
    float qv[HD];
#pragma unroll
    for (int d = 0; d < HD; ++d) qv[d] = qp[d];
    const float scale = 0.35355339059327373f;  // 1/sqrt(8)
    float mx = -1e30f;
    for (int j = 0; j < LL; ++j) {
        float s;
        if (j < len) {
            const float* kp = qkv + ((size_t)(b * LL + j)) * (3 * TD) + TD + hh * HD;
            s = 0.f;
#pragma unroll
            for (int d = 0; d < HD; ++d) s += qv[d] * kp[d];
            s *= scale;
        } else {
            s = -1e9f;
        }
        sc[q * LL + j] = s;
        mx = fmaxf(mx, s);
    }
    float sum = 0.f;
    for (int j = 0; j < LL; ++j) {
        float e = __expf(sc[q * LL + j] - mx);
        sc[q * LL + j] = e;
        sum += e;
    }
    float inv = 1.f / sum;
    float acc[HD];
#pragma unroll
    for (int d = 0; d < HD; ++d) acc[d] = 0.f;
    for (int j = 0; j < LL; ++j) {
        float a = sc[q * LL + j] * inv;
        const float* vp = qkv + ((size_t)(b * LL + j)) * (3 * TD) + 2 * TD + hh * HD;
#pragma unroll
        for (int d = 0; d < HD; ++d) acc[d] += a * vp[d];
    }
    float* op = o + ((size_t)(b * LL + q)) * TD + hh * HD;
#pragma unroll
    for (int d = 0; d < HD; ++d) op[d] = acc[d];
}

// ---------------- residual + LayerNorm (one wave per 64-wide row) -----------
__global__ __launch_bounds__(256) void add_ln(
    float* __restrict__ h, const float* __restrict__ r,
    const float* __restrict__ g, const float* __restrict__ b, int rows) {
    int wave = threadIdx.x >> 5, lane = threadIdx.x & 31;
    int row = blockIdx.x * 8 + wave;
    if (row >= rows) return;
    float* hp = h + (size_t)row * TD;
    const float* rp = r + (size_t)row * TD;
    float x0 = hp[lane] + rp[lane];
    float x1 = hp[lane + 32] + rp[lane + 32];
    float s = x0 + x1;
#pragma unroll
    for (int o = 16; o > 0; o >>= 1) s += __shfl_xor(s, o, 32);
    float mean = s * (1.f / 64.f);
    float d0 = x0 - mean, d1 = x1 - mean;
    float v = d0 * d0 + d1 * d1;
#pragma unroll
    for (int o = 16; o > 0; o >>= 1) v += __shfl_xor(v, o, 32);
    float rstd = rsqrtf(v * (1.f / 64.f) + 1e-5f);
    hp[lane]      = d0 * rstd * g[lane]      + b[lane];
    hp[lane + 32] = d1 * rstd * g[lane + 32] + b[lane + 32];
}

// ---------------- GAT kernels ----------------
__device__ inline unsigned f2ord(float f) {
    unsigned u = __float_as_uint(f);
    return (u & 0x80000000u) ? ~u : (u | 0x80000000u);
}
__device__ inline float ord2f(unsigned u) {
    return (u & 0x80000000u) ? __uint_as_float(u & 0x7fffffffu) : __uint_as_float(~u);
}

__global__ __launch_bounds__(256) void gat_prep(
    const float* __restrict__ h, const float* __restrict__ a_s,
    const float* __restrict__ a_d, float* __restrict__ hs, float* __restrict__ hd,
    unsigned* __restrict__ mx, float* __restrict__ sum, float* __restrict__ acc, int n) {
    int i = blockIdx.x * 256 + threadIdx.x;
    if (i >= n) return;
    const float* hp = h + (size_t)i * HID;
    float s = 0.f, d = 0.f;
#pragma unroll
    for (int k = 0; k < HID; ++k) { s += hp[k] * a_s[k]; d += hp[k] * a_d[k]; }
    hs[i] = s; hd[i] = d;
    mx[i] = 0u;                // == encode(very negative)
    sum[i] = 0.f;
    float* ap = acc + (size_t)i * HID;
#pragma unroll
    for (int k = 0; k < HID; ++k) ap[k] = 0.f;
}

__global__ __launch_bounds__(256) void gat_edge_max(
    const int* __restrict__ ei, const float* __restrict__ hs,
    const float* __restrict__ hd, unsigned* __restrict__ mx, int ne, int n) {
    long long e = (long long)blockIdx.x * 256 + threadIdx.x;
    if (e >= (long long)ne + n) return;
    int s = (e < ne) ? ei[e] : (int)(e - ne);
    int d = (e < ne) ? ei[(size_t)ne + e] : (int)(e - ne);
    float x = hs[s] + hd[d];
    x = (x > 0.f) ? x : 0.2f * x;       // leaky_relu 0.2
    atomicMax(mx + d, f2ord(x));
}

__global__ __launch_bounds__(256) void gat_edge_acc(
    const int* __restrict__ ei, const float* __restrict__ hs,
    const float* __restrict__ hd, const float* __restrict__ h,
    const unsigned* __restrict__ mx, float* __restrict__ sum,
    float* __restrict__ acc, int ne, int n) {
    long long idx = (long long)blockIdx.x * 256 + threadIdx.x;
    long long tot = ((long long)ne + n) * HID;
    if (idx >= tot) return;
    long long e = idx >> 6;
    int k = (int)(idx & 63);
    int s = (e < ne) ? ei[e] : (int)(e - ne);
    int d = (e < ne) ? ei[(size_t)ne + e] : (int)(e - ne);
    float x = hs[s] + hd[d];
    x = (x > 0.f) ? x : 0.2f * x;
    float ex = __expf(x - ord2f(mx[d]));
    if (k == 0) atomicAdd(sum + d, ex);
    atomicAdd(acc + (size_t)d * HID + k, ex * h[(size_t)s * HID + k]);
}

__global__ __launch_bounds__(256) void gat_final(
    const float* __restrict__ acc, const float* __restrict__ sum,
    const float* __restrict__ b, float* __restrict__ out, int n) {
    long long idx = (long long)blockIdx.x * 256 + threadIdx.x;
    if (idx >= (long long)n * HID) return;
    int i = (int)(idx >> 6);
    int k = (int)(idx & 63);
    float v = acc[idx] / (sum[i] + 1e-16f) + b[k];
    out[idx] = (v > 0.f) ? v : (expm1f(v));   // elu
}

// ---------------- masked mean pool ----------------
__global__ __launch_bounds__(64) void pool_kernel(
    const float* __restrict__ h, const int* __restrict__ p1_len,
    const int* __restrict__ p2_len, float* __restrict__ pooled) {
    int s = blockIdx.x;
    int k = threadIdx.x;
    int len = (s < BB) ? p1_len[s] : p2_len[s - BB];
    float acc = 0.f;
    for (int t = 0; t < len; ++t)
        acc += h[((size_t)(s * LL + t)) * TD + k];
    pooled[(size_t)s * TD + k] = acc / (float)len;
}

// ---------------- attention fusion of graph/temporal embeddings -------------
__global__ __launch_bounds__(128) void fuse_att(
    const float* __restrict__ g, const float* __restrict__ tout,
    const int* __restrict__ md, const float* __restrict__ att_w,
    const float* __restrict__ att_b, float* __restrict__ f1, float* __restrict__ f2) {
    int i = threadIdx.x;
    if (i >= BB) return;
    for (int side = 0; side < 2; ++side) {
        int node = md[i * 2 + side];
        const float* gv = g + (size_t)node * HID;
        const float* tv = tout + (size_t)(side * BB + i) * HID;
        float z0 = att_b[0], z1 = att_b[1];
        for (int k = 0; k < HID; ++k) {
            z0 += att_w[k] * gv[k] + att_w[HID + k] * tv[k];
            z1 += att_w[2 * HID + k] * gv[k] + att_w[3 * HID + k] * tv[k];
        }
        float m = fmaxf(z0, z1);
        float e0 = __expf(z0 - m), e1 = __expf(z1 - m);
        float inv = 1.f / (e0 + e1);
        float a0 = e0 * inv, a1 = e1 * inv;
        float* f = (side ? f2 : f1) + (size_t)i * HID;
        for (int k = 0; k < HID; ++k) f[k] = a0 * gv[k] + a1 * tv[k];
    }
}

// ---------------- final head: fm1 (relu) -> fm2 -> sigmoid ------------------
__global__ __launch_bounds__(128) void final_head(
    const float* __restrict__ e1, const float* __restrict__ e2,
    const float* __restrict__ mf, const float* __restrict__ fm1_w,
    const float* __restrict__ fm1_b, const float* __restrict__ fm2_w,
    const float* __restrict__ fm2_b, float* __restrict__ out) {
    int i = threadIdx.x;
    if (i >= BB) return;
    float hm[CD];
    for (int j = 0; j < CD; ++j) {
        const float* wr = fm1_w + (size_t)j * (2 * CD + MF);
        float a = fm1_b[j];
        for (int k = 0; k < CD; ++k) a += wr[k] * e1[(size_t)i * CD + k];
        for (int k = 0; k < CD; ++k) a += wr[CD + k] * e2[(size_t)i * CD + k];
        for (int k = 0; k < MF; ++k) a += wr[2 * CD + k] * mf[(size_t)i * MF + k];
        hm[j] = (a > 0.f) ? a : 0.f;
    }
    float z = fm2_b[0];
    for (int j = 0; j < CD; ++j) z += fm2_w[j] * hm[j];
    out[i] = 1.f / (1.f + __expf(-z));
}

// ============================================================================
extern "C" void kernel_launch(void* const* d_in, const int* in_sizes, int n_in,
                              void* d_out, int out_size, void* d_ws, size_t ws_size,
                              hipStream_t stream) {
    (void)in_sizes; (void)n_in; (void)out_size; (void)ws_size;

    const float* x        = (const float*)d_in[0];
    const int*   ei       = (const int*)d_in[1];
    const float* p1_hist  = (const float*)d_in[2];
    const int*   p1_len   = (const int*)d_in[3];
    const float* p2_hist  = (const float*)d_in[4];
    const int*   p2_len   = (const int*)d_in[5];
    const int*   md       = (const int*)d_in[6];
    const float* mfeat    = (const float*)d_in[7];
    // params in dict insertion order
    const float* gat1_w  = (const float*)d_in[8];
    const float* gat1_as = (const float*)d_in[9];
    const float* gat1_ad = (const float*)d_in[10];
    const float* gat1_b  = (const float*)d_in[11];
    const float* gat2_w  = (const float*)d_in[12];
    const float* gat2_as = (const float*)d_in[13];
    const float* gat2_ad = (const float*)d_in[14];
    const float* gat2_b  = (const float*)d_in[15];
    const float* proj_w  = (const float*)d_in[16];
    const float* proj_b  = (const float*)d_in[17];
    const float* in_w    = (const float*)d_in[18];
    const float* in_b    = (const float*)d_in[19];
    const float* out_w   = (const float*)d_in[20];
    const float* out_b   = (const float*)d_in[21];
    const float* ff1_w   = (const float*)d_in[22];
    const float* ff1_b   = (const float*)d_in[23];
    const float* ff2_w   = (const float*)d_in[24];
    const float* ff2_b   = (const float*)d_in[25];
    const float* ln1_g   = (const float*)d_in[26];
    const float* ln1_b   = (const float*)d_in[27];
    const float* ln2_g   = (const float*)d_in[28];
    const float* ln2_b   = (const float*)d_in[29];
    const float* hfc_w   = (const float*)d_in[30];
    const float* hfc_b   = (const float*)d_in[31];
    const float* att_w   = (const float*)d_in[32];
    const float* att_b   = (const float*)d_in[33];
    const float* fcp_w   = (const float*)d_in[34];
    const float* fcp_b   = (const float*)d_in[35];
    const float* fm1_w   = (const float*)d_in[36];
    const float* fm1_b   = (const float*)d_in[37];
    const float* fm2_w   = (const float*)d_in[38];
    const float* fm2_b   = (const float*)d_in[39];

    float* ws = (float*)d_ws;
    float* w_h    = ws;                           // N*64
    float* w_acc  = w_h   + (size_t)NN * HID;     // N*64
    float* w_g    = w_acc + (size_t)NN * HID;     // N*64
    float* w_hs   = w_g   + (size_t)NN * HID;     // N
    float* w_hd   = w_hs  + NN;                   // N
    float* w_mxf  = w_hd  + NN;                   // N (sortable uint)
    float* w_sum  = w_mxf + NN;                   // N
    float* t_h    = w_sum + NN;                   // R*64
    float* t_qkv  = t_h   + (size_t)RR * TD;      // R*192
    float* t_o    = t_qkv + (size_t)RR * 3 * TD;  // R*64
    float* t_f    = t_o   + (size_t)RR * TD;      // R*64
    float* w_pool = t_f   + (size_t)RR * TD;      // S*64
    float* w_tout = w_pool + (size_t)SS * TD;     // S*64
    float* w_f1   = w_tout + (size_t)SS * TD;     // B*64
    float* w_f2   = w_f1  + (size_t)BB * HID;     // B*64
    float* w_e1   = w_f2  + (size_t)BB * HID;     // B*64
    float* w_e2   = w_e1  + (size_t)BB * CD;      // B*64

    const int ET = EE + NN;  // edges + self loops
    const int nodeTiles = (NN / 16) * 4;          // 3125 * 4

    // ---------------- GAT layer 1 ----------------
    gemm_wmma<<<(nodeTiles + 7) / 8, 256, 0, stream>>>(x, gat1_w, nullptr, w_h,
                                                       NN, HID, FIN, 4, nodeTiles);
    gat_prep<<<(NN + 255) / 256, 256, 0, stream>>>(w_h, gat1_as, gat1_ad, w_hs, w_hd,
                                                   (unsigned*)w_mxf, w_sum, w_acc, NN);
    gat_edge_max<<<(ET + 255) / 256, 256, 0, stream>>>(ei, w_hs, w_hd, (unsigned*)w_mxf, EE, NN);
    gat_edge_acc<<<(int)(((long long)ET * HID + 255) / 256), 256, 0, stream>>>(
        ei, w_hs, w_hd, w_h, (const unsigned*)w_mxf, w_sum, w_acc, EE, NN);
    gat_final<<<(int)(((long long)NN * HID + 255) / 256), 256, 0, stream>>>(
        w_acc, w_sum, gat1_b, w_g, NN);

    // ---------------- GAT layer 2 ----------------
    gemm_wmma<<<(nodeTiles + 7) / 8, 256, 0, stream>>>(w_g, gat2_w, nullptr, w_h,
                                                       NN, HID, HID, 4, nodeTiles);
    gat_prep<<<(NN + 255) / 256, 256, 0, stream>>>(w_h, gat2_as, gat2_ad, w_hs, w_hd,
                                                   (unsigned*)w_mxf, w_sum, w_acc, NN);
    gat_edge_max<<<(ET + 255) / 256, 256, 0, stream>>>(ei, w_hs, w_hd, (unsigned*)w_mxf, EE, NN);
    gat_edge_acc<<<(int)(((long long)ET * HID + 255) / 256), 256, 0, stream>>>(
        ei, w_hs, w_hd, w_h, (const unsigned*)w_mxf, w_sum, w_acc, EE, NN);
    gat_final<<<(int)(((long long)NN * HID + 255) / 256), 256, 0, stream>>>(
        w_acc, w_sum, gat2_b, w_g, NN);

    // ---------------- history projection ----------------
    {
        int halfTiles = (BB * LL / 16) * 4;  // 4096
        gemm_wmma<<<(halfTiles + 7) / 8, 256, 0, stream>>>(p1_hist, proj_w, proj_b,
                                                           t_h, BB * LL, TD, FIN, 4, halfTiles);
        gemm_wmma<<<(halfTiles + 7) / 8, 256, 0, stream>>>(p2_hist, proj_w, proj_b,
                                                           t_h + (size_t)BB * LL * TD,
                                                           BB * LL, TD, FIN, 4, halfTiles);
    }

    // ---------------- transformer encoder ----------------
    for (int l = 0; l < NL; ++l) {
        int qkvTiles = (RR / 16) * 12;  // N=192
        gemm_wmma<<<(qkvTiles + 7) / 8, 256, 0, stream>>>(
            t_h, in_w + (size_t)l * 3 * TD * TD, in_b + (size_t)l * 3 * TD,
            t_qkv, RR, 3 * TD, TD, 12, qkvTiles);
        attn_kernel<<<SS * NH, 128, LL * LL * sizeof(float), stream>>>(t_qkv, p1_len, p2_len, t_o);
        int oTiles = (RR / 16) * 4;
        gemm_wmma<<<(oTiles + 7) / 8, 256, 0, stream>>>(
            t_o, out_w + (size_t)l * TD * TD, out_b + (size_t)l * TD,
            t_f, RR, TD, TD, 4, oTiles);
        add_ln<<<RR / 8, 256, 0, stream>>>(t_h, t_f, ln1_g + (size_t)l * TD,
                                           ln1_b + (size_t)l * TD, RR);
        ffn_fused<<<RR / 16, 256, 16 * MIDLD * sizeof(_Float16), stream>>>(
            t_h, ff1_w + (size_t)l * FFD * TD, ff1_b + (size_t)l * FFD,
            ff2_w + (size_t)l * TD * FFD, ff2_b + (size_t)l * TD, t_f);
        add_ln<<<RR / 8, 256, 0, stream>>>(t_h, t_f, ln2_g + (size_t)l * TD,
                                           ln2_b + (size_t)l * TD, RR);
    }

    // ---------------- pool + temporal fc ----------------
    pool_kernel<<<SS, 64, 0, stream>>>(t_h, p1_len, p2_len, w_pool);
    {
        int tTiles = (SS / 16) * 4;  // 64
        gemm_wmma<<<(tTiles + 7) / 8, 256, 0, stream>>>(w_pool, hfc_w, hfc_b, w_tout,
                                                        SS, HID, TD, 4, tTiles);
    }

    // ---------------- fusion + classifier ----------------
    fuse_att<<<1, 128, 0, stream>>>(w_g, w_tout, md, att_w, att_b, w_f1, w_f2);
    {
        int fTiles = (BB / 16) * 4;  // 32
        gemm_wmma<<<(fTiles + 7) / 8, 256, 0, stream>>>(w_f1, fcp_w, fcp_b, w_e1,
                                                        BB, CD, HID, 4, fTiles);
        gemm_wmma<<<(fTiles + 7) / 8, 256, 0, stream>>>(w_f2, fcp_w, fcp_b, w_e2,
                                                        BB, CD, HID, 4, fTiles);
    }
    final_head<<<1, 128, 0, stream>>>(w_e1, w_e2, mfeat, fm1_w, fm1_b, fm2_w, fm2_b,
                                      (float*)d_out);
}
